// Model_35055523070696
// MI455X (gfx1250) — matrix operand order repaired and checked by reading, not compiled, not run on hardware
//
#include <hip/hip_runtime.h>
#include <math.h>

// ---------------------------------------------------------------------------
// Model constants
// ---------------------------------------------------------------------------
#define B_N     16384
#define COLS    32
#define E_DIM   16
#define ACC     512      // COLS*E = K of GEMM1
#define NE      8
#define SBH     256      // N of GEMM1 (after fce fold), K of GEMM2
#define SBO     128      // N of GEMM2
#define TH_DIM  16
#define T_TASKS 2

#define KT1 (ACC/32)     // 16 K-tiles, GEMM1
#define NT1 (SBH/16)     // 16 N-tiles, GEMM1
#define KT2 (SBH/32)     // 8  K-tiles, GEMM2
#define NT2 (SBO/16)     // 8  N-tiles, GEMM2

#define MT  4            // M-tiles (16 rows each) per wave in both GEMMs

typedef __attribute__((ext_vector_type(16))) _Float16 v16h;
typedef __attribute__((ext_vector_type(8)))  float    v8f;
typedef __attribute__((ext_vector_type(4)))  unsigned int v4u;
typedef __attribute__((ext_vector_type(4)))  float    v4f;

union FragU { v16h h; v4u u4[2]; };
union H2    { unsigned int u; _Float16 h[2]; };

__device__ __forceinline__ float sigmoidf_(float x) { return 1.0f / (1.0f + __expf(-x)); }

// ---------------------------------------------------------------------------
// Prep: W1g[k=0..511][c=0..255] = sum_n fce_w[n] * clip(hardconcrete(u1),0,1)*w1
// packed as f16 pairs into WMMA-B fragment layout (32x16 tiles).
// one thread per (k-pair, c): 65536 threads.
// ---------------------------------------------------------------------------
__global__ __launch_bounds__(256) void k_prep_w1(const float* __restrict__ u1,
                                                 const float* __restrict__ w1,
                                                 const float* __restrict__ alpha1,
                                                 const float* __restrict__ fce_w,
                                                 unsigned int* __restrict__ frag1) {
  int t = blockIdx.x * 256 + threadIdx.x;
  if (t >= (ACC / 2) * SBH) return;
  int kp = t >> 8;          // k-pair 0..255
  int c  = t & 255;         // 0..255
  float la = __logf(alpha1[0]) * (1.0f / 0.9f);
  H2 out;
  #pragma unroll
  for (int i = 0; i < 2; ++i) {
    int k = kp * 2 + i;
    float acc = 0.0f;
    #pragma unroll
    for (int n = 0; n < NE; ++n) {
      int idx = (k * NE + n) * SBH + c;
      float u  = u1[idx];
      float s  = sigmoidf_(__logf(u) - __logf(1.0f - u) + la);
      float s_ = s * 2.1f - 0.1f;
      float z  = fminf(fmaxf(s_, 0.0f), 1.0f);
      acc += fce_w[n] * z * w1[idx];
    }
    out.h[i] = (_Float16)acc;
  }
  int k0    = kp * 2;
  int ktile = k0 >> 5, kin = k0 & 31;
  int ntile = c >> 4,  nin = c & 15;
  int lane  = nin + ((kin >> 4) << 4);
  int j     = (kin & 15) >> 1;
  frag1[(((ntile * KT1 + ktile) * 32 + lane) << 3) + j] = out.u;
}

// ---------------------------------------------------------------------------
// Prep: W2eff[k=0..255][c=0..127] = clip(hardconcrete(u2),0,1)*w2, fragment pack.
// ---------------------------------------------------------------------------
__global__ __launch_bounds__(256) void k_prep_w2(const float* __restrict__ u2,
                                                 const float* __restrict__ w2,
                                                 const float* __restrict__ alpha2,
                                                 unsigned int* __restrict__ frag2) {
  int t = blockIdx.x * 256 + threadIdx.x;
  if (t >= (SBH / 2) * SBO) return;
  int kp = t >> 7;          // 0..127
  int c  = t & 127;
  float la = __logf(alpha2[0]) * (1.0f / 0.9f);
  H2 out;
  #pragma unroll
  for (int i = 0; i < 2; ++i) {
    int k = kp * 2 + i;
    int idx = k * SBO + c;
    float u  = u2[idx];
    float s  = sigmoidf_(__logf(u) - __logf(1.0f - u) + la);
    float s_ = s * 2.1f - 0.1f;
    out.h[i] = (_Float16)(fminf(fmaxf(s_, 0.0f), 1.0f) * w2[idx]);
  }
  int k0    = kp * 2;
  int ktile = k0 >> 5, kin = k0 & 31;
  int ntile = c >> 4,  nin = c & 15;
  int lane  = nin + ((kin >> 4) << 4);
  int j     = (kin & 15) >> 1;
  frag2[(((ntile * KT2 + ktile) * 32 + lane) << 3) + j] = out.u;
}

// ---------------------------------------------------------------------------
// Regularizer count 1: rows of s1_ are [ACC*NE][SBH]; contribution 1 - neg//8.
// s1_ < 0  <=>  u < sigmoid(logit(1/21) - la);  logit(1/21) = -log(20).
// ---------------------------------------------------------------------------
__global__ void k_count1(const float* __restrict__ u1,
                         const float* __restrict__ alpha1,
                         int* __restrict__ cnt) {
  __shared__ int sdata[256];
  float la  = __logf(alpha1[0]) * (1.0f / 0.9f);
  float thr = sigmoidf_(-2.99573227355f - la);
  int acc = 0;
  for (int r = threadIdx.x; r < ACC * NE; r += 256) {
    const float* row = u1 + r * SBH;
    int neg = 0;
    for (int c = 0; c < SBH; ++c) neg += (row[c] < thr) ? 1 : 0;
    acc += 1 - (neg >> 3);          // 1 - negcnt // NE
  }
  sdata[threadIdx.x] = acc;
  __syncthreads();
  for (int s = 128; s > 0; s >>= 1) {
    if ((int)threadIdx.x < s) sdata[threadIdx.x] += sdata[threadIdx.x + s];
    __syncthreads();
  }
  if (threadIdx.x == 0) cnt[0] = sdata[0];
}

// count 2 (rows [SBH][SBO], contribution 1 - neg//128) + final regul write.
__global__ void k_count2(const float* __restrict__ u2,
                         const float* __restrict__ alpha2,
                         const int* __restrict__ cnt,
                         float* __restrict__ out) {
  __shared__ int sdata[256];
  float la  = __logf(alpha2[0]) * (1.0f / 0.9f);
  float thr = sigmoidf_(-2.99573227355f - la);
  const float* row = u2 + threadIdx.x * SBO;
  int neg = 0;
  for (int c = 0; c < SBO; ++c) neg += (row[c] < thr) ? 1 : 0;
  sdata[threadIdx.x] = 1 - (neg >> 7);   // 1 - negcnt // SBO
  __syncthreads();
  for (int s = 128; s > 0; s >>= 1) {
    if ((int)threadIdx.x < s) sdata[threadIdx.x] += sdata[threadIdx.x + s];
    __syncthreads();
  }
  if (threadIdx.x == 0)
    out[T_TASKS * B_N] = 1e-4f * (float)(cnt[0] + sdata[0]);
}

// ---------------------------------------------------------------------------
// Embedding gather -> fp16 xf [B][512]. One thread per (b, col).
// Row = 64B: 4x b128 loads, 2x b128 stores.
// ---------------------------------------------------------------------------
__global__ __launch_bounds__(256) void k_gather(const int* __restrict__ x,
                                                const float* __restrict__ emb,
                                                _Float16* __restrict__ xf) {
  int t = blockIdx.x * 256 + threadIdx.x;
  if (t >= B_N * COLS) return;
  int b = t >> 5, col = t & 31;
  const v4f* src = (const v4f*)(emb + (unsigned)x[t] * E_DIM);
  v4f f0 = src[0], f1 = src[1], f2 = src[2], f3 = src[3];
  H2 h[8];
  #pragma unroll
  for (int i = 0; i < 4; ++i) { h[i*2+0].h[0] = (_Float16)((i==0?f0:i==1?f1:i==2?f2:f3)[0]);
                                h[i*2+0].h[1] = (_Float16)((i==0?f0:i==1?f1:i==2?f2:f3)[1]);
                                h[i*2+1].h[0] = (_Float16)((i==0?f0:i==1?f1:i==2?f2:f3)[2]);
                                h[i*2+1].h[1] = (_Float16)((i==0?f0:i==1?f1:i==2?f2:f3)[3]); }
  v4u* d = (v4u*)(xf + b * ACC + col * E_DIM);
  v4u o0 = { h[0].u, h[1].u, h[2].u, h[3].u };
  v4u o1 = { h[4].u, h[5].u, h[6].u, h[7].u };
  d[0] = o0; d[1] = o1;
}

// ---------------------------------------------------------------------------
// GEMM1: g[16384,256] = xf[16384,512] @ W1g   (fp16 in, fp32 acc, fp16 out)
// wave = 4 M-tiles (64 rows) x 128 cols (8 N-tiles): 32 accumulators (256 VGPR)
// B fragments double-buffered in registers; each B load feeds 4 WMMAs.
// Single-wave workgroups for maximum WGP spread: 512 blocks x 32 threads.
// ---------------------------------------------------------------------------
__global__ __launch_bounds__(32) void k_gemm1(const _Float16* __restrict__ xf,
                                              const unsigned int* __restrict__ frag1,
                                              _Float16* __restrict__ g) {
  int wid    = blockIdx.x;             // 0..511
  int lane   = threadIdx.x & 31;
  int mblock = wid >> 1;               // 0..255 (64 rows each)
  int nstrip = wid & 1;                // 0..1 (each strip = 8 N-tiles)
  int kgrp   = lane >> 4;
  const _Float16* arow = xf + (mblock * 64 + (lane & 15)) * ACC;

  v8f acc[MT][8] = {};
  #pragma unroll
  for (int kt = 0; kt < KT1; ++kt) {
    FragU a[MT];
    #pragma unroll
    for (int m = 0; m < MT; ++m) {
      const _Float16* ap = arow + m * 16 * ACC + kt * 32 + kgrp * 16;
      a[m].u4[0] = *(const v4u*)ap;
      a[m].u4[1] = *(const v4u*)(ap + 8);
    }
    const unsigned int* bbase =
        frag1 + (((nstrip * 8) * KT1 + kt) * 32 + lane) * 8;
    FragU bcur;
    bcur.u4[0] = *(const v4u*)bbase;
    bcur.u4[1] = *(const v4u*)(bbase + 4);
    #pragma unroll
    for (int j = 0; j < 8; ++j) {
      FragU bnext;
      if (j < 7) {                      // prefetch next N-tile's fragment
        const unsigned int* bp = bbase + (j + 1) * (KT1 * 32 * 8);
        bnext.u4[0] = *(const v4u*)bp;
        bnext.u4[1] = *(const v4u*)(bp + 4);
      }
      #pragma unroll
      for (int m = 0; m < MT; ++m)
        acc[m][j] = __builtin_amdgcn_wmma_f32_16x16x32_f16(
            false, a[m].h, false, bcur.h, (short)0, acc[m][j], false, false);
      if (j < 7) bcur = bnext;
    }
  }
  // D layout: lane holds N=lane%16, rows M = 8*(lane/16)+r
  int n0 = (nstrip << 7) + (lane & 15);
  #pragma unroll
  for (int m = 0; m < MT; ++m) {
    int m0 = mblock * 64 + m * 16 + ((lane >> 4) << 3);
    #pragma unroll
    for (int j = 0; j < 8; ++j) {
      int nc = n0 + j * 16;
      #pragma unroll
      for (int r = 0; r < 8; ++r)
        g[(m0 + r) * SBH + nc] = (_Float16)acc[m][j][r];
    }
  }
}

// ---------------------------------------------------------------------------
// GEMM2: r[16384,128] = relu(g[16384,256] @ W2eff + fce_b)
// wave = 4 M-tiles (64 rows) x all 128 cols, K loop of 8.
// Single-wave workgroups: 256 blocks x 32 threads.
// ---------------------------------------------------------------------------
__global__ __launch_bounds__(32) void k_gemm2(const _Float16* __restrict__ g,
                                              const unsigned int* __restrict__ frag2,
                                              const float* __restrict__ fce_b,
                                              float* __restrict__ rbuf) {
  int wid    = blockIdx.x;              // 0..255
  int lane   = threadIdx.x & 31;
  int mblock = wid;
  int kgrp   = lane >> 4;
  const _Float16* arow = g + (mblock * 64 + (lane & 15)) * SBH;
  float bias = fce_b[0];

  v8f acc[MT][NT2] = {};
  #pragma unroll
  for (int kt = 0; kt < KT2; ++kt) {
    FragU a[MT];
    #pragma unroll
    for (int m = 0; m < MT; ++m) {
      const _Float16* ap = arow + m * 16 * SBH + kt * 32 + kgrp * 16;
      a[m].u4[0] = *(const v4u*)ap;
      a[m].u4[1] = *(const v4u*)(ap + 8);
    }
    const unsigned int* bbase = frag2 + (kt * 32 + lane) * 8;
    FragU bcur;
    bcur.u4[0] = *(const v4u*)bbase;
    bcur.u4[1] = *(const v4u*)(bbase + 4);
    #pragma unroll
    for (int j = 0; j < NT2; ++j) {
      FragU bnext;
      if (j < NT2 - 1) {
        const unsigned int* bp = bbase + (j + 1) * (KT2 * 32 * 8);
        bnext.u4[0] = *(const v4u*)bp;
        bnext.u4[1] = *(const v4u*)(bp + 4);
      }
      #pragma unroll
      for (int m = 0; m < MT; ++m)
        acc[m][j] = __builtin_amdgcn_wmma_f32_16x16x32_f16(
            false, a[m].h, false, bcur.h, (short)0, acc[m][j], false, false);
      if (j < NT2 - 1) bcur = bnext;
    }
  }
  int n0 = lane & 15;
  #pragma unroll
  for (int m = 0; m < MT; ++m) {
    int m0 = mblock * 64 + m * 16 + ((lane >> 4) << 3);
    #pragma unroll
    for (int j = 0; j < NT2; ++j) {
      int nc = n0 + j * 16;
      #pragma unroll
      for (int r = 0; r < 8; ++r)
        rbuf[(m0 + r) * SBO + nc] = fmaxf(acc[m][j][r] + bias, 0.0f);
    }
  }
}

// ---------------------------------------------------------------------------
// Towers: out[t*B + b] = sigmoid( relu(r[b]@tw1[t]+tb1[t]) @ tw2[t] + tb2[t] )
// ---------------------------------------------------------------------------
__global__ __launch_bounds__(256) void k_towers(const float* __restrict__ rbuf,
                                                const float* __restrict__ tw1,
                                                const float* __restrict__ tb1,
                                                const float* __restrict__ tw2,
                                                const float* __restrict__ tb2,
                                                float* __restrict__ out) {
  __shared__ float s_tw1[T_TASKS * SBO * TH_DIM];   // 16 KB
  __shared__ float s_tb1[T_TASKS * TH_DIM];
  __shared__ float s_tw2[T_TASKS * TH_DIM];
  __shared__ float s_tb2[T_TASKS];
  for (int i = threadIdx.x; i < T_TASKS * SBO * TH_DIM; i += 256) s_tw1[i] = tw1[i];
  if (threadIdx.x < T_TASKS * TH_DIM) {
    s_tb1[threadIdx.x] = tb1[threadIdx.x];
    s_tw2[threadIdx.x] = tw2[threadIdx.x];
  }
  if (threadIdx.x < T_TASKS) s_tb2[threadIdx.x] = tb2[threadIdx.x];
  __syncthreads();

  int b = blockIdx.x * 256 + threadIdx.x;
  if (b >= B_N) return;
  const float* row = rbuf + b * SBO;
  #pragma unroll
  for (int t = 0; t < T_TASKS; ++t) {
    float h[TH_DIM];
    #pragma unroll
    for (int j = 0; j < TH_DIM; ++j) h[j] = s_tb1[t * TH_DIM + j];
    for (int d = 0; d < SBO; ++d) {
      float rv = row[d];
      const float* wrow = &s_tw1[(t * SBO + d) * TH_DIM];
      #pragma unroll
      for (int j = 0; j < TH_DIM; ++j) h[j] = fmaf(rv, wrow[j], h[j]);
    }
    float o = s_tb2[t];
    #pragma unroll
    for (int j = 0; j < TH_DIM; ++j) o = fmaf(fmaxf(h[j], 0.0f), s_tw2[t * TH_DIM + j], o);
    out[t * B_N + b] = sigmoidf_(o);
  }
}

// ---------------------------------------------------------------------------
// Launch
// ---------------------------------------------------------------------------
extern "C" void kernel_launch(void* const* d_in, const int* in_sizes, int n_in,
                              void* d_out, int out_size, void* d_ws, size_t ws_size,
                              hipStream_t stream) {
  const int*   x      = (const int*)d_in[0];
  const float* emb    = (const float*)d_in[1];
  const float* alpha1 = (const float*)d_in[2];
  const float* u1     = (const float*)d_in[3];
  const float* w1     = (const float*)d_in[4];
  const float* alpha2 = (const float*)d_in[5];
  const float* u2     = (const float*)d_in[6];
  const float* w2     = (const float*)d_in[7];
  const float* fce_w  = (const float*)d_in[8];
  const float* fce_b  = (const float*)d_in[9];
  const float* tw1    = (const float*)d_in[10];
  const float* tb1    = (const float*)d_in[11];
  const float* tw2    = (const float*)d_in[12];
  const float* tb2    = (const float*)d_in[13];
  float* out = (float*)d_out;

  char* ws = (char*)d_ws;
  size_t off = 0;
  unsigned int* frag1 = (unsigned int*)(ws + off); off += (size_t)NT1 * KT1 * 32 * 8 * 4;   // 256 KB
  unsigned int* frag2 = (unsigned int*)(ws + off); off += (size_t)NT2 * KT2 * 32 * 8 * 4;   //  64 KB
  off = (off + 255) & ~(size_t)255;
  _Float16* xf   = (_Float16*)(ws + off); off += (size_t)B_N * ACC * 2;                      //  16 MB
  _Float16* gbuf = (_Float16*)(ws + off); off += (size_t)B_N * SBH * 2;                      //   8 MB
  float*    rbuf = (float*)(ws + off);    off += (size_t)B_N * SBO * 4;                      //   8 MB
  int*      cnt  = (int*)(ws + off);

  k_prep_w1<<<(ACC / 2 * SBH + 255) / 256, 256, 0, stream>>>(u1, w1, alpha1, fce_w, frag1);
  k_prep_w2<<<(SBH / 2 * SBO + 255) / 256, 256, 0, stream>>>(u2, w2, alpha2, frag2);
  k_count1<<<1, 256, 0, stream>>>(u1, alpha1, cnt);
  k_gather<<<(B_N * COLS + 255) / 256, 256, 0, stream>>>(x, emb, xf);
  k_gemm1<<<(B_N / 64) * 2, 32, 0, stream>>>(xf, frag1, gbuf);        // 512 single-wave WGs
  k_gemm2<<<(B_N / 64), 32, 0, stream>>>(gbuf, frag2, fce_b, rbuf);   // 256 single-wave WGs
  k_towers<<<(B_N + 255) / 256, 256, 0, stream>>>(rbuf, tw1, tb1, tw2, tb2, out);
  k_count2<<<1, 256, 0, stream>>>(u2, alpha2, cnt, out);
}
